// SparseBasisSelector_27650999452118
// MI455X (gfx1250) — matrix operand order repaired and checked by reading, not compiled, not run on hardware
//
#include <hip/hip_runtime.h>
#include <hip/hip_bf16.h>

// ---------------------------------------------------------------------------
// Problem constants (from reference)
// ---------------------------------------------------------------------------
#define BATCH   16
#define NSEQ    1024
#define TLEN    512
#define CCH     32
#define KK8     8
#define L1OUT   253   // (512-8)/2+1
#define L2OUT   123   // (253-8)/2+1
#define L3OUT   58    // (123-8)/2+1
#define INDIM   1856  // 32*58
#define TOPK    256
#define TPB     128

typedef __attribute__((ext_vector_type(16))) _Float16 v16h;
typedef __attribute__((ext_vector_type(8)))  float    v8f;
typedef __attribute__((ext_vector_type(4)))  unsigned v4u;

// ---------------------------------------------------------------------------
// WMMA fragment loaders (wave32 16-bit; this file's contiguous k = 16*hi + s order, same for A and B)
// A (16x32 f16): lane L (M = L&15), VGPR0-3 hold K = hi*16 + [0..7],
//                VGPR4-7 hold K = hi*16 + 8 + [0..7], hi = L>>4.
// Ws is row-major [32][256] f16 -> both halves are contiguous 16B chunks.
// ---------------------------------------------------------------------------
__device__ __forceinline__ v16h load_a_frag(const _Float16* Ws, int m, int kc, int lane) {
  const int row = m + (lane & 15);
  const int hi8 = (lane >> 4) * 16;
  union { v16h h; v4u u[2]; } ua;
  ua.u[0] = *(const v4u*)(Ws + row * 256 + kc * 32 + hi8);
  ua.u[1] = *(const v4u*)(Ws + row * 256 + kc * 32 + 8 + hi8);
  return ua.h;
}

// B (32x16 f16): lane L (N = L&15), VGPR v holds rows K = hi*16 + 2v, 2v+1
// (low half = even K), hi = L>>4.  B is im2col of src[c][2n+k], src row-major
// with stride `strideH` halves.  K index (kc*32 + hi*16 + 2v) is always even,
// so k = K&7 is even -> the f16 pair (k, k+1) stays inside one channel row
// and the byte address is dword aligned -> single aligned b32 LDS load.
__device__ __forceinline__ v16h load_b_frag(const _Float16* src, int strideH,
                                            int tn, int kc, int lane) {
  const int n    = tn * 16 + (lane & 15);
  const int hi16 = (lane >> 4) * 16;
  const unsigned* p32 = (const unsigned*)src;
  union { v16h h; unsigned u[8]; } ub;
#pragma unroll
  for (int v = 0; v < 8; ++v) {
    const int kk  = kc * 32 + hi16 + 2 * v;  // even
    const int cin = kk >> 3;
    const int k   = kk & 7;                  // even
    ub.u[v] = p32[(cin * strideH + 2 * n + k) >> 1];
  }
  return ub.h;
}

// One conv layer as GEMM: D(32 x nValid) = relu(W(32x256) * im2col(src) + bias)
// nTiles = ceil(nValid/16); 2*nTiles WMMA tiles split over 4 waves (uniform).
__device__ __forceinline__ void gemm_layer(const _Float16* Ws, const _Float16* src,
                                           int srcStride, const float* bias,
                                           int nTiles, int nValid,
                                           _Float16* dstH, float* dstF, int dstStride,
                                           int tid) {
  const int wave = tid >> 5;
  const int lane = tid & 31;
  const int tilesTotal = nTiles * 2;
  for (int tq = wave; tq < tilesTotal; tq += 4) {   // uniform trip count per wave
    const int m  = (tq & 1) * 16;
    const int tn = tq >> 1;
    v8f c = {};
#pragma unroll
    for (int kc = 0; kc < 8; ++kc) {
      v16h a = load_a_frag(Ws, m, kc, lane);
      v16h b = load_b_frag(src, srcStride, tn, kc, lane);
      c = __builtin_amdgcn_wmma_f32_16x16x32_f16(false, a, false, b,
                                                 (short)0, c, false, false);
    }
    // C/D layout: VGPR r -> row m + (lane>>4)*8 + r, col = tn*16 + (lane&15)
    const int col   = tn * 16 + (lane & 15);
    const int rbase = m + (lane >> 4) * 8;
    if (col < nValid) {
#pragma unroll
      for (int r = 0; r < 8; ++r) {
        float v = fmaxf(c[r] + bias[rbase + r], 0.0f);
        if (dstH) dstH[(rbase + r) * dstStride + col] = (_Float16)v;
        else      dstF[(rbase + r) * dstStride + col] = v;
      }
    }
  }
}

// ---------------------------------------------------------------------------
// Kernel 1: per-signal conv stack + scoring.  One block per (b,n) signal.
// ---------------------------------------------------------------------------
__global__ __launch_bounds__(TPB) void conv_score_kernel(
    const float* __restrict__ h,
    const float* __restrict__ W0, const float* __restrict__ b0,
    const float* __restrict__ W1, const float* __restrict__ b1,
    const float* __restrict__ W2, const float* __restrict__ b2,
    const float* __restrict__ Wp, const float* __restrict__ bp,
    float* __restrict__ scores) {
  __shared__ __align__(16) float    xs[TLEN];
  __shared__            float       W0s[CCH * KK8];
  __shared__            float       b0s[CCH], b1s[CCH], b2s[CCH];
  __shared__ __align__(16) _Float16 W1s[CCH * 256];
  __shared__ __align__(16) _Float16 W2s[CCH * 256];
  __shared__ __align__(16) _Float16 out0s[CCH * 256];  // valid t < 253
  __shared__ __align__(16) _Float16 out1s[CCH * 128];  // valid t < 123
  __shared__ __align__(16) float    out2s[CCH * 64];   // valid t < 58
  __shared__            float       red[TPB];

  const int tid = threadIdx.x;
  const int sig = blockIdx.x;
  const float* xrow = h + (size_t)sig * TLEN;

  for (int i = tid; i < TLEN; i += TPB) xs[i] = xrow[i];
  for (int i = tid; i < CCH * KK8; i += TPB) W0s[i] = W0[i];
  if (tid < CCH) { b0s[tid] = b0[tid]; b1s[tid] = b1[tid]; b2s[tid] = b2[tid]; }
  for (int i = tid; i < CCH * 256; i += TPB) {
    W1s[i] = (_Float16)W1[i];
    W2s[i] = (_Float16)W2[i];
  }
  __syncthreads();

  // Layer 0 (1 input channel) in VALU: out0[c][t] = relu(b0 + sum_k W0[c,k]*x[2t+k])
  for (int o = tid; o < CCH * L1OUT; o += TPB) {
    const int c = o & 31;
    const int t = o >> 5;
    float acc = b0s[c];
#pragma unroll
    for (int k = 0; k < KK8; ++k) acc += W0s[c * KK8 + k] * xs[2 * t + k];
    out0s[c * 256 + t] = (_Float16)fmaxf(acc, 0.0f);
  }
  __syncthreads();

  // Layer 1: 32x123 = relu(W1[32x256] * im2col(out0) + b1)   (WMMA)
  gemm_layer(W1s, out0s, 256, b1s, /*nTiles=*/8, /*nValid=*/L2OUT,
             out1s, nullptr, 128, tid);
  __syncthreads();

  // Layer 2: 32x58 = relu(W2[32x256] * im2col(out1) + b2)    (WMMA)
  gemm_layer(W2s, out1s, 128, b2s, /*nTiles=*/4, /*nValid=*/L3OUT,
             nullptr, out2s, 64, tid);
  __syncthreads();

  // Projection: score = sigmoid(bp + sum_{c,t} out2[c][t] * Wp[c*58+t])
  float acc = 0.0f;
  for (int i = tid; i < INDIM; i += TPB) {
    const int c = i / L3OUT;
    const int t = i - c * L3OUT;
    acc += out2s[c * 64 + t] * Wp[i];
  }
  red[tid] = acc;
  __syncthreads();
  if (tid == 0) {
    float s = bp[0];
    for (int i = 0; i < TPB; ++i) s += red[i];
    scores[sig] = 1.0f / (1.0f + expf(-s));
  }
}

// ---------------------------------------------------------------------------
// Kernel 2: per-batch top-256 (bitonic sort of packed keys) + scaled gather.
// Key = (score_bits << 32) | (1023 - idx): sigmoid > 0 so float bits are
// monotonic as uint; descending u64 sort == descending score, ties -> lower
// original index first (matches jax.lax.top_k).
// ---------------------------------------------------------------------------
__global__ __launch_bounds__(512) void topk_gather_kernel(
    const float* __restrict__ h, const float* __restrict__ scores,
    float* __restrict__ out_h, int* __restrict__ out_idx) {
  __shared__ unsigned long long keys[NSEQ];
  const int b   = blockIdx.x;
  const int tid = threadIdx.x;

  for (int e = tid; e < NSEQ; e += 512) {
    const float s = scores[b * NSEQ + e];
    keys[e] = ((unsigned long long)__float_as_uint(s) << 32) |
              (unsigned)(NSEQ - 1 - e);
  }
  __syncthreads();

  for (int k = 2; k <= NSEQ; k <<= 1) {
    for (int j = k >> 1; j > 0; j >>= 1) {
      for (int e = tid; e < NSEQ; e += 512) {
        const int ixj = e ^ j;
        if (ixj > e) {
          const unsigned long long x = keys[e];
          const unsigned long long y = keys[ixj];
          const bool up = ((e & k) == 0);          // up blocks -> descending
          const bool sw = up ? (y > x) : (x > y);
          if (sw) { keys[e] = y; keys[ixj] = x; }
        }
      }
      __syncthreads();
    }
  }

  // keys[0..255] are the top-k in final order; gather rows of h scaled by value
  for (int j = 0; j < TOPK; ++j) {
    const unsigned long long key = keys[j];
    const int   idx = (NSEQ - 1) - (int)(key & 0xffffffffull);
    const float val = __uint_as_float((unsigned)(key >> 32));
    out_h[((size_t)b * TOPK + j) * TLEN + tid] =
        h[((size_t)b * NSEQ + idx) * TLEN + tid] * val;
  }
  for (int j = tid; j < TOPK; j += 512)
    out_idx[b * TOPK + j] = (NSEQ - 1) - (int)(keys[j] & 0xffffffffull);
}

// ---------------------------------------------------------------------------
// Launch
// ---------------------------------------------------------------------------
extern "C" void kernel_launch(void* const* d_in, const int* in_sizes, int n_in,
                              void* d_out, int out_size, void* d_ws, size_t ws_size,
                              hipStream_t stream) {
  const float* h  = (const float*)d_in[0];
  const float* W0 = (const float*)d_in[1];
  const float* b0 = (const float*)d_in[2];
  const float* W1 = (const float*)d_in[3];
  const float* b1 = (const float*)d_in[4];
  const float* W2 = (const float*)d_in[5];
  const float* b2 = (const float*)d_in[6];
  const float* Wp = (const float*)d_in[7];
  const float* bp = (const float*)d_in[8];

  float* scores  = (float*)d_ws;                       // BATCH*NSEQ floats
  float* out_h   = (float*)d_out;                      // (16,256,512) f32
  int*   out_idx = (int*)(out_h + (size_t)BATCH * TOPK * TLEN);  // (16,256,1) i32

  conv_score_kernel<<<BATCH * NSEQ, TPB, 0, stream>>>(
      h, W0, b0, W1, b1, W2, b2, Wp, bp, scores);
  topk_gather_kernel<<<BATCH, 512, 0, stream>>>(h, scores, out_h, out_idx);
}
